// TransBlockV2_50208167690491
// MI455X (gfx1250) — matrix operand order repaired and checked by reading, not compiled, not run on hardware
//
#include <hip/hip_runtime.h>
#include <hip/hip_bf16.h>
#include <stdint.h>

// Problem constants (match reference)
#define C_DIM   128
#define H_DIM   512
#define NCOEF   25
#define GCOLS   6144      // 2 eq experts * 5 l * 512  +  2 ne experts * 512
// LDS strides (padded to break bank conflicts; *2B stays 16B aligned)
#define XSTR    136
#define HSTR    520
#define GSTR    520
#define SXELEMS (16 * XSTR)

typedef __attribute__((ext_vector_type(16))) __bf16 v16bf;
typedef __attribute__((ext_vector_type(8)))  float  v8f;

__device__ __forceinline__ float silu_f(float x) {
  return x / (1.0f + __expf(-x));
}

__device__ __forceinline__ v8f wmma_bf16(v16bf a, v16bf b, v8f c) {
  // D = A(16x32) * B(32x16) + C(16x16), f32 accum
  return __builtin_amdgcn_wmma_f32_16x16x32_bf16(false, a, false, b, (short)0, c, false, false);
}

// CDNA5 async copy: 16B global -> LDS per lane, tracked by ASYNCcnt.
__device__ __forceinline__ void async_copy_b128(uint32_t lds_byte_addr, const void* gsrc) {
  asm volatile("global_load_async_to_lds_b128 %0, %1, off"
               :
               : "v"(lds_byte_addr), "v"((unsigned long long)(uintptr_t)gsrc)
               : "memory");
}
__device__ __forceinline__ void wait_async0() {
  asm volatile("s_wait_asynccnt 0" ::: "memory");
}

// A fragment (16x32, M rows striped over lanes) from row-major LDS tile.
// ISA 16-bit A layout: lanes 0-15 hold K {0..7, 16..23}; lanes 16-31 hold K {8..15, 24..31}.
__device__ __forceinline__ v16bf load_a_frag(const __bf16* base, int stride, int kbase, int lane) {
  int row = lane & 15;
  int kb  = (lane < 16) ? 0 : 8;
  const __bf16* p = base + row * stride + kbase + kb;
  union { v16bf v; uint4 q[2]; } u;
  u.q[0] = *(const uint4*)(p);
  u.q[1] = *(const uint4*)(p + 16);
  return u.v;
}

// B fragment (32x16, N cols striped over lanes) from transposed weights Wt[n][k] (contiguous K).
// ISA 16-bit B layout: lanes 0-15 hold K {0..7, 16..23}; lanes 16-31 hold K {8..15, 24..31}.
__device__ __forceinline__ v16bf load_b_frag(const __bf16* wt, int ktot, int ncol_base, int kbase, int lane) {
  int col = lane & 15;
  int kb  = (lane < 16) ? 0 : 8;
  const __bf16* p = wt + (size_t)(ncol_base + col) * ktot + kbase + kb;
  union { v16bf v; uint4 q[2]; } u;
  u.q[0] = *(const uint4*)(p);
  u.q[1] = *(const uint4*)(p + 16);
  return u.v;
}

__device__ __forceinline__ int l_of_s(int s) {
  return (s == 0) ? 0 : (s < 4) ? 1 : (s < 9) ? 2 : (s < 16) ? 3 : 4;
}

// Block-collective: gate tile (16 nodes x 512 cols) = silu(cTile @ GWt[:,gcol..+512] + gb) -> bf16 LDS.
// Callers place __syncthreads() around uses.
__device__ __forceinline__ void compute_gate_tile(const __bf16* sC, const __bf16* GWt,
                                                  const float* gbias, int gcol_base,
                                                  __bf16* dst, int wave, int lane) {
  v8f acc0 = 0.f, acc1 = 0.f, acc2 = 0.f, acc3 = 0.f;
  #pragma unroll
  for (int kk = 0; kk < C_DIM; kk += 32) {
    v16bf a = load_a_frag(sC, XSTR, kk, lane);
    acc0 = wmma_bf16(a, load_b_frag(GWt, C_DIM, gcol_base + wave * 64 +  0, kk, lane), acc0);
    acc1 = wmma_bf16(a, load_b_frag(GWt, C_DIM, gcol_base + wave * 64 + 16, kk, lane), acc1);
    acc2 = wmma_bf16(a, load_b_frag(GWt, C_DIM, gcol_base + wave * 64 + 32, kk, lane), acc2);
    acc3 = wmma_bf16(a, load_b_frag(GWt, C_DIM, gcol_base + wave * 64 + 48, kk, lane), acc3);
  }
  int colhalf = lane & 15;
  int rowoff  = (lane < 16) ? 0 : 8;
  v8f accs[4] = {acc0, acc1, acc2, acc3};
  #pragma unroll
  for (int t = 0; t < 4; ++t) {
    int col = wave * 64 + t * 16 + colhalf;
    float gb = gbias[gcol_base + col];
    #pragma unroll
    for (int v = 0; v < 8; ++v) {
      float x = accs[t][v] + gb;
      dst[(v + rowoff) * GSTR + col] = (__bf16)silu_f(x);
    }
  }
}

// ---------------------------------------------------------------------------
// Edge aggregation: c_sum[dst] += t_ij[e]; cnt[dst] += 1
// ---------------------------------------------------------------------------
__global__ __launch_bounds__(256) void edge_accum_kernel(const float* __restrict__ t_ij,
                                                         const int* __restrict__ edge_index,
                                                         float* __restrict__ c_sum,
                                                         float* __restrict__ cnt, int E) {
  int gid = blockIdx.x * 256 + threadIdx.x;   // E * 32 lanes, 4 channels each
  int e = gid >> 5;
  if (e >= E) return;
  int c4 = (gid & 31) * 4;
  int dst = edge_index[E + e];                // edge_index[1][e]
  const float4 v = *(const float4*)(t_ij + (size_t)e * C_DIM + c4);
  float* d = c_sum + (size_t)dst * C_DIM + c4;
  atomicAdd(d + 0, v.x);
  atomicAdd(d + 1, v.y);
  atomicAdd(d + 2, v.z);
  atomicAdd(d + 3, v.w);
  if ((gid & 31) == 0) atomicAdd(&cnt[dst], 1.0f);
}

__global__ __launch_bounds__(256) void finalize_c_kernel(float* __restrict__ c_sum,
                                                         const float* __restrict__ cnt, int N) {
  int i = blockIdx.x * 256 + threadIdx.x;
  if (i >= N * C_DIM) return;
  float d = fmaxf(cnt[i >> 7], 1.0f);
  c_sum[i] = c_sum[i] / d;                    // in place: c_sum becomes c_i
}

// ---------------------------------------------------------------------------
// Router: w = softmax(silu(c_i@rW1+rb1)@rW2+rb2)   (one block of 128 per node)
// ---------------------------------------------------------------------------
__global__ __launch_bounds__(128) void router_kernel(const float* __restrict__ c_i,
                                                     const float* __restrict__ rW1,
                                                     const float* __restrict__ rb1,
                                                     const float* __restrict__ rW2,
                                                     const float* __restrict__ rb2,
                                                     float* __restrict__ wprob, int N) {
  __shared__ float sc[128];
  __shared__ float sa[128];
  __shared__ float slg[4];
  int n = blockIdx.x;
  int t = threadIdx.x;
  sc[t] = c_i[(size_t)n * C_DIM + t];
  __syncthreads();
  float acc = rb1[t];
  for (int c = 0; c < 128; ++c) acc += sc[c] * rW1[c * 128 + t];
  sa[t] = silu_f(acc);
  __syncthreads();
  if (t < 4) {
    float lg = rb2[t];
    for (int c = 0; c < 128; ++c) lg += sa[c] * rW2[c * 4 + t];
    slg[t] = lg;
  }
  __syncthreads();
  if (t == 0) {
    float mx = fmaxf(fmaxf(slg[0], slg[1]), fmaxf(slg[2], slg[3]));
    float e0 = __expf(slg[0] - mx), e1 = __expf(slg[1] - mx);
    float e2 = __expf(slg[2] - mx), e3 = __expf(slg[3] - mx);
    float inv = 1.0f / (e0 + e1 + e2 + e3);
    wprob[(size_t)n * 4 + 0] = e0 * inv;
    wprob[(size_t)n * 4 + 1] = e1 * inv;
    wprob[(size_t)n * 4 + 2] = e2 * inv;
    wprob[(size_t)n * 4 + 3] = e3 * inv;
  }
}

// ---------------------------------------------------------------------------
// Prep: x_emb f32 -> bf16 copy (enables async b128 tile staging in main kernel)
// ---------------------------------------------------------------------------
__global__ __launch_bounds__(256) void prep_xb_kernel(const float* __restrict__ x,
                                                      __bf16* __restrict__ xb, int total4) {
  int i = blockIdx.x * 256 + threadIdx.x;
  if (i >= total4) return;
  float4 v = ((const float4*)x)[i];
  __bf16* d = xb + (size_t)i * 4;
  d[0] = (__bf16)v.x;
  d[1] = (__bf16)v.y;
  d[2] = (__bf16)v.z;
  d[3] = (__bf16)v.w;
}

// ---------------------------------------------------------------------------
// Weight prep: f32 -> bf16, transposed to Wt[n][k] (K contiguous)
// ---------------------------------------------------------------------------
__global__ __launch_bounds__(256) void prep_w1t_kernel(const float* __restrict__ eq_W1,
                                                       const float* __restrict__ ne_W1,
                                                       __bf16* __restrict__ W1t) {
  int idx = blockIdx.x * 256 + threadIdx.x;           // e*65536 + h*128 + c
  if (idx >= 4 * H_DIM * C_DIM) return;
  int e = idx >> 16, rem = idx & 65535;
  int h = rem >> 7, c = rem & 127;
  const float* src = (e < 2) ? (eq_W1 + (size_t)e * 65536) : (ne_W1 + (size_t)(e - 2) * 65536);
  W1t[idx] = (__bf16)src[c * H_DIM + h];
}

__global__ __launch_bounds__(256) void prep_w2t_kernel(const float* __restrict__ eq_W2,
                                                       const float* __restrict__ ne_W2,
                                                       __bf16* __restrict__ W2t) {
  int idx = blockIdx.x * 256 + threadIdx.x;           // e*65536 + c*512 + h
  if (idx >= 4 * C_DIM * H_DIM) return;
  int e = idx >> 16, rem = idx & 65535;
  int c = rem >> 9, h = rem & 511;
  const float* src = (e < 2) ? (eq_W2 + (size_t)e * 65536) : (ne_W2 + (size_t)(e - 2) * 65536);
  W2t[idx] = (__bf16)src[h * C_DIM + c];
}

__global__ __launch_bounds__(256) void prep_gwt_kernel(const float* __restrict__ eq_gW,
                                                       const float* __restrict__ eq_gb,
                                                       const float* __restrict__ ne_gW,
                                                       const float* __restrict__ ne_gb,
                                                       __bf16* __restrict__ GWt,
                                                       float* __restrict__ gbias) {
  int idx = blockIdx.x * 256 + threadIdx.x;           // j*128 + c
  if (idx >= GCOLS * C_DIM) return;
  int j = idx >> 7, c = idx & 127;
  float v, b;
  if (j < 5120) {                                     // eq gates: expert k, col within 2560
    int k = j / 2560, col = j % 2560;
    v = eq_gW[(size_t)k * 327680 + (size_t)c * 2560 + col];
    b = eq_gb[k * 2560 + col];
  } else {
    int j2 = j - 5120;
    int k = j2 >> 9, col = j2 & 511;
    v = ne_gW[(size_t)k * 65536 + (size_t)c * 512 + col];
    b = ne_gb[k * 512 + col];
  }
  GWt[idx] = (__bf16)v;
  if (c == 0) gbias[j] = b;
}

// ---------------------------------------------------------------------------
// Fused expert kernel: one block = 16 nodes, 8 waves (256 threads).
// For each s (25): GEMM1 (K=128) -> act*gate -> GEMM2 (K=512) -> w-scaled accumulate.
// x tiles staged to LDS with async b128 copies, double-buffered across s.
// ---------------------------------------------------------------------------
__global__ __launch_bounds__(256) void moe_main_kernel(const __bf16* __restrict__ xb,
                                                       const float* __restrict__ c_i,
                                                       const float* __restrict__ wprob,
                                                       const __bf16* __restrict__ W1t,
                                                       const __bf16* __restrict__ W2t,
                                                       const __bf16* __restrict__ GWt,
                                                       const float* __restrict__ gbias,
                                                       const float* __restrict__ ne_b1,
                                                       const float* __restrict__ ne_b2,
                                                       float* __restrict__ out, int N) {
  __shared__ __bf16 sX[2][SXELEMS];     // double-buffered x tile (16 nodes x 128), bf16
  __shared__ __bf16 sC[16 * XSTR];      // c_i tile (16 nodes x 128), bf16
  __shared__ __bf16 sH[16 * HSTR];      // activated hidden (16 x 512), bf16
  __shared__ __bf16 sGeq[2][16 * GSTR]; // eq gates for current l (2 experts x 16 x 512)
  __shared__ __bf16 sGne[2][16 * GSTR]; // ne gates (2 experts x 16 x 512)
  __shared__ float  sW[64];             // router weights (16 nodes x 4)

  const int n0   = blockIdx.x * 16;
  const int tid  = threadIdx.x;
  const int wave = tid >> 5;
  const int lane = tid & 31;
  const int colhalf = lane & 15;
  const int rowoff  = (lane < 16) ? 0 : 8;

  // Per-thread x-tile copy slot: row r, 8-element chunk c8 (16B) -> one async b128 per thread.
  const int xr  = tid >> 4;
  const int xc8 = (tid & 15) * 8;
  const int xnr = min(n0 + xr, N - 1);
  const __bf16* xsrc_base = xb + (size_t)xnr * NCOEF * C_DIM + xc8;
  const uint32_t sx_lds0 = (uint32_t)(uintptr_t)(&sX[0][0]) + (uint32_t)((xr * XSTR + xc8) * 2);

  // Kick off async staging of the s=0 tile immediately (overlaps ne-gate WMMAs below).
  async_copy_b128(sx_lds0, xsrc_base + 0 * C_DIM);

  // Load c_i tile -> bf16 LDS (each thread: one row-chunk of 8)
  {
    const float* src = c_i + (size_t)xnr * C_DIM + xc8;
    #pragma unroll
    for (int i = 0; i < 8; ++i) sC[xr * XSTR + xc8 + i] = (__bf16)src[i];
  }
  if (tid < 64) {
    int nr = min(n0 + (tid >> 2), N - 1);
    sW[tid] = wprob[(size_t)nr * 4 + (tid & 3)];
  }
  __syncthreads();

  // ne gates: constant over s (computed once) — overlaps the in-flight async x load
  compute_gate_tile(sC, GWt, gbias, 5120, sGne[0], wave, lane);
  compute_gate_tile(sC, GWt, gbias, 5632, sGne[1], wave, lane);

  int cur_l = -1;
  for (int s = 0; s < NCOEF; ++s) {
    const int l = l_of_s(s);

    // Recompute eq gate tiles when l changes (before the publish barrier)
    if (l != cur_l) {
      compute_gate_tile(sC, GWt, gbias, 0 * 2560 + l * H_DIM, sGeq[0], wave, lane);
      compute_gate_tile(sC, GWt, gbias, 1 * 2560 + l * H_DIM, sGeq[1], wave, lane);
      cur_l = l;
    }

    wait_async0();       // this wave's async tile for s is resident
    __syncthreads();     // publish x tile + gate tiles to the whole block

    // Prefetch next s tile into the other buffer while we compute on this one.
    if (s + 1 < NCOEF) {
      async_copy_b128(sx_lds0 + (uint32_t)(((s + 1) & 1) * SXELEMS * 2),
                      xsrc_base + (size_t)(s + 1) * C_DIM);
    }
    const __bf16* sXc = &sX[s & 1][0];

    v8f outacc = 0.f;   // this wave's output col-tile [wave*16 .. wave*16+15]

    for (int e = 0; e < 4; ++e) {
      const bool is_ne = (e >= 2);

      // ---- GEMM1: (16 x 128) @ (128 x 512); this wave covers cols wave*64..+63
      const __bf16* W1 = W1t + (size_t)e * H_DIM * C_DIM;
      v8f a0 = 0.f, a1 = 0.f, a2 = 0.f, a3 = 0.f;
      #pragma unroll
      for (int kk = 0; kk < C_DIM; kk += 32) {
        v16bf af = load_a_frag(sXc, XSTR, kk, lane);
        a0 = wmma_bf16(af, load_b_frag(W1, C_DIM, wave * 64 +  0, kk, lane), a0);
        a1 = wmma_bf16(af, load_b_frag(W1, C_DIM, wave * 64 + 16, kk, lane), a1);
        a2 = wmma_bf16(af, load_b_frag(W1, C_DIM, wave * 64 + 32, kk, lane), a2);
        a3 = wmma_bf16(af, load_b_frag(W1, C_DIM, wave * 64 + 48, kk, lane), a3);
      }

      // ---- Activation * gate -> bf16 LDS
      const __bf16* gtile = is_ne ? sGne[e - 2] : sGeq[e];
      const bool do_silu = is_ne || (l == 0);
      v8f accs[4] = {a0, a1, a2, a3};
      #pragma unroll
      for (int t = 0; t < 4; ++t) {
        int col = wave * 64 + t * 16 + colhalf;
        float b1v = is_ne ? ne_b1[(e - 2) * H_DIM + col] : 0.0f;
        #pragma unroll
        for (int v = 0; v < 8; ++v) {
          int row = v + rowoff;
          float x = accs[t][v] + b1v;
          float y = do_silu ? silu_f(x) : x;
          float g = (float)gtile[row * GSTR + col];
          sH[row * HSTR + col] = (__bf16)(y * g);
        }
      }
      __syncthreads();

      // ---- GEMM2: (16 x 512) @ (512 x 128); this wave covers cols wave*16..+15
      const __bf16* W2 = W2t + (size_t)e * C_DIM * H_DIM;
      v8f acc2 = 0.f;
      #pragma unroll
      for (int kk = 0; kk < H_DIM; kk += 32) {
        v16bf af = load_a_frag(sH, HSTR, kk, lane);
        acc2 = wmma_bf16(af, load_b_frag(W2, H_DIM, wave * 16, kk, lane), acc2);
      }

      // ---- router-weighted accumulate (+ b2 for ne experts)
      float b2v = is_ne ? ne_b2[(e - 2) * C_DIM + wave * 16 + colhalf] : 0.0f;
      #pragma unroll
      for (int v = 0; v < 8; ++v) {
        int row = v + rowoff;
        float wv = sW[row * 4 + e];
        outacc[v] += wv * (acc2[v] + b2v);
      }
      __syncthreads();   // before next expert overwrites sH
    }

    // ---- store out tile (f32)
    #pragma unroll
    for (int v = 0; v < 8; ++v) {
      int row = v + rowoff;
      if (n0 + row < N)
        out[((size_t)(n0 + row) * NCOEF + s) * C_DIM + wave * 16 + colhalf] = outacc[v];
    }
    // No trailing barrier needed: x is double-buffered and the intra-expert
    // barriers already order all sH / sGeq reuse across the block.
  }
}

// ---------------------------------------------------------------------------
// Host launcher
// ---------------------------------------------------------------------------
extern "C" void kernel_launch(void* const* d_in, const int* in_sizes, int n_in,
                              void* d_out, int out_size, void* d_ws, size_t ws_size,
                              hipStream_t stream) {
  const float* x_emb  = (const float*)d_in[0];
  const float* t_ij   = (const float*)d_in[1];
  const int*   eidx   = (const int*)  d_in[2];
  const float* rW1    = (const float*)d_in[3];
  const float* rb1    = (const float*)d_in[4];
  const float* rW2    = (const float*)d_in[5];
  const float* rb2    = (const float*)d_in[6];
  const float* eq_W1  = (const float*)d_in[7];
  const float* eq_gW  = (const float*)d_in[8];
  const float* eq_gb  = (const float*)d_in[9];
  const float* eq_W2  = (const float*)d_in[10];
  const float* ne_W1  = (const float*)d_in[11];
  const float* ne_b1  = (const float*)d_in[12];
  const float* ne_gW  = (const float*)d_in[13];
  const float* ne_gb  = (const float*)d_in[14];
  const float* ne_W2  = (const float*)d_in[15];
  const float* ne_b2  = (const float*)d_in[16];

  const int N = in_sizes[0] / (NCOEF * C_DIM);
  const int E = in_sizes[1] / C_DIM;

  // Workspace carve-up (256B aligned)
  uint8_t* w8 = (uint8_t*)d_ws;
  size_t off = 0;
  auto wsAlloc = [&](size_t bytes) -> void* {
    void* p = (void*)(w8 + off);
    off += (bytes + 255) & ~(size_t)255;
    return p;
  };
  float*  c_sum = (float*) wsAlloc((size_t)N * C_DIM * sizeof(float));  // becomes c_i
  float*  cnt   = (float*) wsAlloc((size_t)N * sizeof(float));
  float*  wprob = (float*) wsAlloc((size_t)N * 4 * sizeof(float));
  __bf16* W1t   = (__bf16*)wsAlloc((size_t)4 * H_DIM * C_DIM * sizeof(__bf16));
  __bf16* W2t   = (__bf16*)wsAlloc((size_t)4 * C_DIM * H_DIM * sizeof(__bf16));
  __bf16* GWt   = (__bf16*)wsAlloc((size_t)GCOLS * C_DIM * sizeof(__bf16));
  float*  gbias = (float*) wsAlloc((size_t)GCOLS * sizeof(float));
  __bf16* xbf   = (__bf16*)wsAlloc((size_t)N * NCOEF * C_DIM * sizeof(__bf16));
  (void)ws_size; (void)n_in; (void)out_size;

  hipMemsetAsync(c_sum, 0, (size_t)N * C_DIM * sizeof(float), stream);
  hipMemsetAsync(cnt,   0, (size_t)N * sizeof(float), stream);

  // Prep passes (bf16 conversions / transposes)
  const int xtotal4 = (N * NCOEF * C_DIM) / 4;
  prep_xb_kernel<<<(xtotal4 + 255) / 256, 256, 0, stream>>>(x_emb, xbf, xtotal4);
  prep_w1t_kernel<<<(4 * H_DIM * C_DIM + 255) / 256, 256, 0, stream>>>(eq_W1, ne_W1, W1t);
  prep_w2t_kernel<<<(4 * C_DIM * H_DIM + 255) / 256, 256, 0, stream>>>(eq_W2, ne_W2, W2t);
  prep_gwt_kernel<<<(GCOLS * C_DIM + 255) / 256, 256, 0, stream>>>(eq_gW, eq_gb, ne_gW, ne_gb, GWt, gbias);

  // Edge mean aggregation -> c_i
  edge_accum_kernel<<<((size_t)E * 32 + 255) / 256, 256, 0, stream>>>(t_ij, eidx, c_sum, cnt, E);
  finalize_c_kernel<<<(N * C_DIM + 255) / 256, 256, 0, stream>>>(c_sum, cnt, N);

  // Router softmax weights
  router_kernel<<<N, 128, 0, stream>>>(c_sum, rW1, rb1, rW2, rb2, wprob, N);

  // Fused WMMA expert kernel
  moe_main_kernel<<<(N + 15) / 16, 256, 0, stream>>>(xbf, c_sum, wprob, W1t, W2t, GWt, gbias,
                                                     ne_b1, ne_b2, (float*)d_out, N);
}
